// GAT_43559558316174
// MI455X (gfx1250) — matrix operand order repaired and checked by reading, not compiled, not run on hardware
//
#include <hip/hip_runtime.h>
#include <cstddef>

// ---------------------------------------------------------------------------
// GAT on MI455X (gfx1250).
//  * All GEMMs: v_wmma_f32_16x16x32_bf16 (wave32), fp32 accumulate.
//  * All GEMM operands pre-converted to bf16; B operands pre-transposed to
//    [N,K] so every tile stage is a straight byte copy -> async load-to-LDS
//    (ASYNCcnt) when the toolchain exposes the gfx1250 builtins.
//  * Block tile 128x128, K-step 64 (two WMMA k-chunks per LDS stage),
//    8 waves, each wave = 4x2 grid of 16x16 WMMA accumulators.
// ---------------------------------------------------------------------------

typedef __attribute__((ext_vector_type(16))) __bf16 v16bf;
typedef __attribute__((ext_vector_type(8)))  float  v8f;
typedef __attribute__((ext_vector_type(4)))  int    v4i_t;

#define NV     4096
#define FDIM   512
#define ODIM   512
#define HEADS  8
#define ALPHA  0.2f
#define NEGBIG -9.0e15f

#if defined(__HIP_DEVICE_COMPILE__) && \
    __has_builtin(__builtin_amdgcn_global_load_async_to_lds_b128) && \
    __has_builtin(__builtin_amdgcn_s_wait_asynccnt)
#define USE_ASYNC_LDS 1
#else
#define USE_ASYNC_LDS 0
#endif

#define AS3 __attribute__((address_space(3)))

__device__ __forceinline__ unsigned short f2bf(float f) {
  unsigned u = __builtin_bit_cast(unsigned, f);
  u += 0x7FFFu + ((u >> 16) & 1u);           // round-to-nearest-even
  return (unsigned short)(u >> 16);
}

// 16-byte global -> LDS copy (async on gfx1250 toolchains that expose it)
__device__ __forceinline__ void copy16_g2l(void* lds, const void* g) {
#if USE_ASYNC_LDS
  __builtin_amdgcn_global_load_async_to_lds_b128((v4i_t*)g, (AS3 v4i_t*)lds, 0, 0);
#else
  *(uint4*)lds = *(const uint4*)g;
#endif
}
__device__ __forceinline__ void stage_wait() {
#if USE_ASYNC_LDS
  __builtin_amdgcn_s_wait_asynccnt(0);
#endif
  __syncthreads();
}

enum { EPI_F32 = 0, EPI_BIAS_BF16 = 1, EPI_ELU_BF16 = 2, EPI_MASK = 3 };

// LDS row stride: 64 bf16 = 32 words data + 4 pad = 36 words (144B, 16B-mult)
#define SAW 36

union FragU { v16bf v; unsigned u[8]; };

// C[M,N] = A[M,K] @ B^T  with A bf16 row-major [M,K], B bf16 row-major [N,K].
// Block tile 128x128, K-step 64. 256 threads = 8 waves; wave grid 2(M)x4(N);
// each wave: 4x2 16x16 tiles (64x32 of C).
template<int EPI>
__global__ __launch_bounds__(256)
void gemm_bf16nt_kernel(const unsigned short* __restrict__ A, int lda,
                        const unsigned short* __restrict__ B, int ldb,
                        void* __restrict__ Cv, int ldc, int K,
                        const float* __restrict__ bias,
                        long bStride, long cStrideBytes)
{
  __shared__ unsigned ldsA[128 * SAW];
  __shared__ unsigned ldsB[128 * SAW];

  B  += (long)blockIdx.z * bStride;
  Cv  = (char*)Cv + (long)blockIdx.z * cStrideBytes;

  const int tid  = threadIdx.x;
  const int lane = tid & 31;
  const int wv   = tid >> 5;
  const int wr   = wv & 1;       // M sub-tile: 64*wr
  const int wc   = wv >> 1;      // N sub-tile: 32*wc
  const int hlf  = lane >> 4;
  const int mn   = lane & 15;

  const int m0 = blockIdx.y * 128;
  const int n0 = blockIdx.x * 128;

  v8f acc[4][2] = {};

  for (int k0 = 0; k0 < K; k0 += 64) {
    // stage 128x64 bf16 tiles of A and B (row = 128B = 8 x 16B chunks)
#pragma unroll
    for (int i = 0; i < 4; ++i) {
      int c   = i * 256 + tid;          // 0..1023 chunks
      int row = c >> 3, cb = c & 7;
      copy16_g2l((char*)ldsA + row * (SAW * 4) + cb * 16,
                 (const char*)(A + (size_t)(m0 + row) * lda + k0) + cb * 16);
      copy16_g2l((char*)ldsB + row * (SAW * 4) + cb * 16,
                 (const char*)(B + (size_t)(n0 + row) * ldb + k0) + cb * 16);
    }
    stage_wait();

#pragma unroll
    for (int kc = 0; kc < 2; ++kc) {
      FragU fa[4], fb[2];
#pragma unroll
      for (int mi = 0; mi < 4; ++mi) {
        const unsigned* rowp = &ldsA[(wr * 64 + mi * 16 + mn) * SAW + kc * 16];
#pragma unroll
        for (int r = 0; r < 8; ++r) {
          int w = (r < 4) ? (r + 4 * hlf) : (r + 4 + 4 * hlf);  // A K-interleave
          fa[mi].u[r] = rowp[w];
        }
      }
#pragma unroll
      for (int ni = 0; ni < 2; ++ni) {
        const unsigned* rowp = &ldsB[(wc * 32 + ni * 16 + mn) * SAW + kc * 16];
#pragma unroll
        for (int r = 0; r < 8; ++r)
          fb[ni].u[r] = rowp[(r < 4) ? (r + 4 * hlf) : (r + 4 + 4 * hlf)];  // B K-interleave (same as A)
      }
#pragma unroll
      for (int mi = 0; mi < 4; ++mi)
#pragma unroll
        for (int ni = 0; ni < 2; ++ni)
          acc[mi][ni] = __builtin_amdgcn_wmma_f32_16x16x32_bf16(
              false, fa[mi].v, false, fb[ni].v, (short)0, acc[mi][ni], false, false);
    }
    __syncthreads();
  }

  // epilogue (C/D layout: lanes 0-15 N=lane,M=r; lanes 16-31 N=lane-16,M=8+r)
#pragma unroll
  for (int mi = 0; mi < 4; ++mi) {
#pragma unroll
    for (int ni = 0; ni < 2; ++ni) {
#pragma unroll
      for (int r = 0; r < 8; ++r) {
        int row = m0 + wr * 64 + mi * 16 + hlf * 8 + r;
        int col = n0 + wc * 32 + ni * 16 + mn;
        float v = acc[mi][ni][r];
        if (EPI == EPI_MASK) {
          ((unsigned char*)Cv)[(size_t)row * ldc + col] = (v != 0.0f) ? 1 : 0;
        } else if (EPI == EPI_F32) {
          ((float*)Cv)[(size_t)row * ldc + col] = v;
        } else if (EPI == EPI_BIAS_BF16) {
          v += bias[col];
          ((unsigned short*)Cv)[(size_t)row * ldc + col] = f2bf(v);
        } else {  // EPI_ELU_BF16
          v = v > 0.f ? v : (__expf(v) - 1.f);
          ((unsigned short*)Cv)[(size_t)row * ldc + col] = f2bf(v);
        }
      }
    }
  }
}

// ---------------------------------------------------------------------------
// fp32 [R,C] -> bf16 transposed [C,R]
__global__ __launch_bounds__(256)
void transpose_to_bf16_kernel(const float* __restrict__ in, int ldin,
                              unsigned short* __restrict__ out, int ldout,
                              long inStride, long outStride)
{
  __shared__ float tile[32][33];
  in  += (long)blockIdx.z * inStride;
  out += (long)blockIdx.z * outStride;
  int tx = threadIdx.x & 31, ty = threadIdx.x >> 5;
  int r0 = blockIdx.y * 32, c0 = blockIdx.x * 32;
#pragma unroll
  for (int i = 0; i < 4; ++i)
    tile[ty + 8 * i][tx] = in[(size_t)(r0 + ty + 8 * i) * ldin + c0 + tx];
  __syncthreads();
#pragma unroll
  for (int i = 0; i < 4; ++i)
    out[(size_t)(c0 + ty + 8 * i) * ldout + r0 + tx] = f2bf(tile[tx][ty + 8 * i]);
}

// fp32 -> bf16 elementwise
__global__ __launch_bounds__(256)
void cvt_bf16_kernel(const float* __restrict__ in, unsigned short* __restrict__ out, long n)
{
  long i = (long)blockIdx.x * blockDim.x + threadIdx.x;
  long stride = (long)gridDim.x * blockDim.x;
  for (; i < n; i += stride) out[i] = f2bf(in[i]);
}

// ---------------------------------------------------------------------------
__device__ __forceinline__ float block_reduce_max(float v, float* s) {
  int t = threadIdx.x;
  s[t] = v; __syncthreads();
  for (int off = 128; off > 0; off >>= 1) {
    if (t < off) s[t] = fmaxf(s[t], s[t + off]);
    __syncthreads();
  }
  float r = s[0]; __syncthreads();
  return r;
}
__device__ __forceinline__ float block_reduce_sum(float v, float* s) {
  int t = threadIdx.x;
  s[t] = v; __syncthreads();
  for (int off = 128; off > 0; off >>= 1) {
    if (t < off) s[t] += s[t + off];
    __syncthreads();
  }
  float r = s[0]; __syncthreads();
  return r;
}

// s_src[i] = Wh[i,:]·a[0:O],  s_dst[i] = Wh[i,:]·a[O:2O]
__global__ __launch_bounds__(256)
void dual_dot_kernel(const float* __restrict__ Wh, const float* __restrict__ a,
                     float* __restrict__ ssrc, float* __restrict__ sdst)
{
  __shared__ float s1[256], s2[256];
  int row = blockIdx.x, t = threadIdx.x;
  const float* wrow = Wh + (size_t)row * ODIM;
  float a1 = 0.f, a2 = 0.f;
  for (int j = t; j < ODIM; j += 256) {
    float x = wrow[j];
    a1 += x * a[j];
    a2 += x * a[ODIM + j];
  }
  s1[t] = a1; s2[t] = a2; __syncthreads();
  for (int off = 128; off > 0; off >>= 1) {
    if (t < off) { s1[t] += s1[t + off]; s2[t] += s2[t + off]; }
    __syncthreads();
  }
  if (t == 0) { ssrc[row] = s1[0]; sdst[row] = s2[0]; }
}

// att[i,:] = softmax_j(mask ? leaky_relu(s_src[i]+s_dst[j]) : -9e15), bf16 out
__global__ __launch_bounds__(256)
void masked_softmax_kernel(const float* __restrict__ ssrc, const float* __restrict__ sdst,
                           const unsigned char* __restrict__ mask,
                           unsigned short* __restrict__ att)
{
  __shared__ float red[256];
  int row = blockIdx.x, t = threadIdx.x;
  size_t base = (size_t)row * NV;
  float si = ssrc[row];

  float vmax = -3.4e38f;
  for (int j = t; j < NV; j += 256) {
    float e = si + sdst[j];
    e = e > 0.f ? e : ALPHA * e;
    e = mask[base + j] ? e : NEGBIG;
    vmax = fmaxf(vmax, e);
  }
  vmax = block_reduce_max(vmax, red);

  float ssum = 0.f;
  for (int j = t; j < NV; j += 256) {
    float e = si + sdst[j];
    e = e > 0.f ? e : ALPHA * e;
    e = mask[base + j] ? e : NEGBIG;
    ssum += __expf(e - vmax);       // all-masked row -> uniform (matches ref)
  }
  ssum = block_reduce_sum(ssum, red);
  float inv = 1.0f / ssum;

  for (int j = t; j < NV; j += 256) {
    float e = si + sdst[j];
    e = e > 0.f ? e : ALPHA * e;
    e = mask[base + j] ? e : NEGBIG;
    att[base + j] = f2bf(__expf(e - vmax) * inv);
  }
}

// out[i,:] = log_softmax(elu(in[i,:]))
__global__ __launch_bounds__(256)
void elu_logsoftmax_kernel(const float* __restrict__ in, float* __restrict__ out)
{
  __shared__ float red[256];
  int row = blockIdx.x, t = threadIdx.x;
  const float* x = in + (size_t)row * ODIM;
  float* y = out + (size_t)row * ODIM;

  float vmax = -3.4e38f;
  for (int j = t; j < ODIM; j += 256) {
    float v = x[j];
    v = v > 0.f ? v : (__expf(v) - 1.f);
    y[j] = v;
    vmax = fmaxf(vmax, v);
  }
  vmax = block_reduce_max(vmax, red);
  float ssum = 0.f;
  for (int j = t; j < ODIM; j += 256) ssum += __expf(y[j] - vmax);
  ssum = block_reduce_sum(ssum, red);
  float lse = vmax + __logf(ssum);
  for (int j = t; j < ODIM; j += 256) y[j] = y[j] - lse;
}

// ---------------------------------------------------------------------------
extern "C" void kernel_launch(void* const* d_in, const int* in_sizes, int n_in,
                              void* d_out, int out_size, void* d_ws, size_t ws_size,
                              hipStream_t stream)
{
  (void)in_sizes; (void)n_in; (void)out_size; (void)ws_size;

  const float* x_in    = (const float*)d_in[0];
  const float* wq_w    = (const float*)d_in[1];
  const float* wq_b    = (const float*)d_in[2];
  const float* wk_w    = (const float*)d_in[3];
  const float* wk_b    = (const float*)d_in[4];
  const float* W_heads = (const float*)d_in[5];
  const float* a_heads = (const float*)d_in[6];
  const float* W_out   = (const float*)d_in[7];
  const float* a_out   = (const float*)d_in[8];

  char* wsp = (char*)d_ws;
  size_t off = 0;
  auto alloc = [&](size_t bytes) -> char* {
    char* p = wsp + off;
    off = (off + bytes + 255) & ~(size_t)255;
    return p;
  };
  typedef unsigned short ushort_t;
  ushort_t* x_bf   = (ushort_t*)alloc((size_t)NV * FDIM * 2);
  ushort_t* wqT    = (ushort_t*)alloc((size_t)FDIM * FDIM * 2);
  ushort_t* wkT    = (ushort_t*)alloc((size_t)FDIM * FDIM * 2);
  ushort_t* WhdT   = (ushort_t*)alloc((size_t)HEADS * FDIM * ODIM * 2);
  ushort_t* WoutT  = (ushort_t*)alloc((size_t)ODIM * (HEADS * ODIM) * 2);
  ushort_t* qx_bf  = (ushort_t*)alloc((size_t)NV * FDIM * 2);
  ushort_t* kx_bf  = (ushort_t*)alloc((size_t)NV * FDIM * 2);
  float*    Wh_all = (float*)   alloc((size_t)HEADS * NV * ODIM * 4);
  ushort_t* WhT    = (ushort_t*)alloc((size_t)ODIM * NV * 2);     // reused per head
  ushort_t* att_bf = (ushort_t*)alloc((size_t)NV * NV * 2);
  ushort_t* x_cat  = (ushort_t*)alloc((size_t)NV * (HEADS * ODIM) * 2);
  float*    WhO    = (float*)   alloc((size_t)NV * ODIM * 4);
  ushort_t* WhOT   = (ushort_t*)alloc((size_t)ODIM * NV * 2);
  float*    outp   = (float*)   alloc((size_t)NV * ODIM * 4);
  float*    ssrc   = (float*)   alloc((size_t)NV * 4);
  float*    sdst   = (float*)   alloc((size_t)NV * 4);
  unsigned char* mask = (unsigned char*)alloc((size_t)NV * NV);

  dim3 blk(256);

  // --- one-time precision conversion / B-operand transposes ---
  cvt_bf16_kernel<<<512, blk, 0, stream>>>(x_in, x_bf, (long)NV * FDIM);
  transpose_to_bf16_kernel<<<dim3(FDIM / 32, FDIM / 32, 1), blk, 0, stream>>>(
      wq_w, FDIM, wqT, FDIM, 0, 0);
  transpose_to_bf16_kernel<<<dim3(FDIM / 32, FDIM / 32, 1), blk, 0, stream>>>(
      wk_w, FDIM, wkT, FDIM, 0, 0);
  transpose_to_bf16_kernel<<<dim3(ODIM / 32, FDIM / 32, HEADS), blk, 0, stream>>>(
      W_heads, ODIM, WhdT, FDIM, (long)FDIM * ODIM, (long)FDIM * ODIM);
  transpose_to_bf16_kernel<<<dim3(ODIM / 32, (HEADS * ODIM) / 32, 1), blk, 0, stream>>>(
      W_out, ODIM, WoutT, HEADS * ODIM, 0, 0);

  // --- qx/kx projections (bias, bf16 output) ---
  gemm_bf16nt_kernel<EPI_BIAS_BF16><<<dim3(FDIM / 128, NV / 128, 1), blk, 0, stream>>>(
      x_bf, FDIM, wqT, FDIM, qx_bf, FDIM, FDIM, wq_b, 0, 0);
  gemm_bf16nt_kernel<EPI_BIAS_BF16><<<dim3(FDIM / 128, NV / 128, 1), blk, 0, stream>>>(
      x_bf, FDIM, wkT, FDIM, kx_bf, FDIM, FDIM, wk_b, 0, 0);

  // --- mask = (qx @ kx^T) != 0  (adj > 0 <=> S != 0; S never materialized) ---
  gemm_bf16nt_kernel<EPI_MASK><<<dim3(NV / 128, NV / 128, 1), blk, 0, stream>>>(
      qx_bf, FDIM, kx_bf, FDIM, mask, NV, FDIM, nullptr, 0, 0);

  // --- Wh_all[h] = x_in @ W_heads[h]  (fp32 out; batched over blockIdx.z) ---
  gemm_bf16nt_kernel<EPI_F32><<<dim3(ODIM / 128, NV / 128, HEADS), blk, 0, stream>>>(
      x_bf, FDIM, WhdT, FDIM, Wh_all, ODIM, FDIM, nullptr,
      (long)FDIM * ODIM, (long)NV * ODIM * 4);

  // --- per-head attention + elu, concat into x_cat ---
  for (int h = 0; h < HEADS; ++h) {
    const float* Wh = Wh_all + (size_t)h * NV * ODIM;
    transpose_to_bf16_kernel<<<dim3(ODIM / 32, NV / 32, 1), blk, 0, stream>>>(
        Wh, ODIM, WhT, NV, 0, 0);
    dual_dot_kernel<<<NV, blk, 0, stream>>>(Wh, a_heads + (size_t)h * 2 * ODIM, ssrc, sdst);
    masked_softmax_kernel<<<NV, blk, 0, stream>>>(ssrc, sdst, mask, att_bf);
    gemm_bf16nt_kernel<EPI_ELU_BF16><<<dim3(ODIM / 128, NV / 128, 1), blk, 0, stream>>>(
        att_bf, NV, WhT, NV, x_cat + (size_t)h * ODIM, HEADS * ODIM, NV, nullptr, 0, 0);
  }

  // --- output GAT layer ---
  gemm_bf16nt_kernel<EPI_F32><<<dim3(ODIM / 128, NV / 128, 1), blk, 0, stream>>>(
      x_cat, HEADS * ODIM, WoutT, HEADS * ODIM, WhO, ODIM, HEADS * ODIM, nullptr, 0, 0);
  transpose_to_bf16_kernel<<<dim3(ODIM / 32, NV / 32, 1), blk, 0, stream>>>(
      WhO, ODIM, WhOT, NV, 0, 0);
  dual_dot_kernel<<<NV, blk, 0, stream>>>(WhO, a_out, ssrc, sdst);
  masked_softmax_kernel<<<NV, blk, 0, stream>>>(ssrc, sdst, mask, att_bf);
  gemm_bf16nt_kernel<EPI_F32><<<dim3(ODIM / 128, NV / 128, 1), blk, 0, stream>>>(
      att_bf, NV, WhOT, NV, outp, ODIM, NV, nullptr, 0, 0);

  // --- elu + log_softmax -> d_out [N, O] fp32 ---
  elu_logsoftmax_kernel<<<NV, blk, 0, stream>>>(outp, (float*)d_out);
}
